// ObjectDetector1D_31748398252283
// MI455X (gfx1250) — matrix operand order repaired and checked by reading, not compiled, not run on hardware
//
#include <hip/hip_runtime.h>
#include <hip/hip_bf16.h>

typedef __attribute__((ext_vector_type(16))) __bf16 v16bf;
typedef __attribute__((ext_vector_type(8)))  float  v8f;

union AFrag { v16bf v; uint4 q[2]; };

#define BATCH 16
#define WAVES_PER_BLK 4

// ---------------------------------------------------------------------------
// Generic conv1d (k in {1,3}, stride in {1,2}) as implicit GEMM on WMMA bf16.
// X: [B][Cin][Lin] bf16.  Wp: [ntaps][CoutPad][Cin] bf16 (prepacked, zero-pad).
// Yb (optional): bf16 out, Yf (optional): f32 out; both [B][Cout][Lout].
// Each wave computes one 16-row Cout tile x NBLK 16-wide position tiles:
// A (weight) fragments are loaded once per (tap, K-chunk) and reused for
// NBLK WMMAs; B fragments come from the LDS-transposed X window staged
// cooperatively by the 4 waves of the block (shared across 4 Cout tiles).
// ---------------------------------------------------------------------------
template <int NBLK>
__global__ __launch_bounds__(128) void conv1d_wmma_bf16(
    const __bf16* __restrict__ X, const __bf16* __restrict__ Wp,
    const float* __restrict__ bias, __bf16* __restrict__ Yb,
    float* __restrict__ Yf,
    int Cin, int Cout, int CoutPad, int Lin, int Lout,
    int stride, int ntaps, int pad, int relu, int nCoTiles, int nLTiles)
{
  __shared__ __align__(16) __bf16 lds[16896]; // >= max(npos*Cin) over all layers
  const int tid  = threadIdx.x;
  const int lane = tid & 31;
  const int wv   = tid >> 5;
  const int nCoBlks = (nCoTiles + WAVES_PER_BLK - 1) / WAVES_PER_BLK;

  int bid = blockIdx.x;
  const int lt = bid % nLTiles;  bid /= nLTiles;
  const int cb = bid % nCoBlks;
  const int b  = bid / nCoBlks;

  const int l0   = lt * (16 * NBLK);
  const int p0   = stride * l0 - pad;                 // first input position
  const int npos = stride * (16 * NBLK - 1) + ntaps;  // window length

  // Cooperative stage: lds[pos][ci] = X[b][ci][p0+pos] (transposed for B-frag)
  const __bf16* Xb = X + (size_t)b * Cin * Lin;
  for (int ci = wv; ci < Cin; ci += WAVES_PER_BLK) {
    const __bf16* xrow = Xb + (size_t)ci * Lin;
    for (int pp = lane; pp < npos; pp += 32) {
      const int gp = p0 + pp;
      __bf16 val = (gp >= 0 && gp < Lin) ? xrow[gp] : (__bf16)0.0f;
      lds[pp * Cin + ci] = val;
    }
  }
  __syncthreads();

  const int ct = cb * WAVES_PER_BLK + wv;
  if (ct >= nCoTiles) return;
  const int co0 = ct * 16;

  // CDNA5 16-bit A layout: lane m=lane&15, K base 8*(lane>>4); elements 0..7
  // at K=kb.., 8..15 at K=kb+16..  B layout: lane n=lane&15, same K order as A.
  const int mA  = lane & 15;
  const int kbA = (lane >> 4) * 8;
  const int nB  = lane & 15;
  const int kbB = (lane >> 4) * 8;
  const int nChunks = Cin >> 5;

  v8f acc[NBLK];
  #pragma unroll
  for (int j = 0; j < NBLK; ++j) acc[j] = {0.f,0.f,0.f,0.f,0.f,0.f,0.f,0.f};

  for (int t = 0; t < ntaps; ++t) {
    const __bf16* wrow = Wp + ((size_t)t * CoutPad + (co0 + mA)) * Cin;
    for (int c = 0; c < nChunks; ++c) {
      AFrag a;
      const int kA = (c << 5) + kbA;
      a.q[0] = *(const uint4*)(wrow + kA);        // global_load_b128 (amortized)
      a.q[1] = *(const uint4*)(wrow + kA + 16);
      const int kB = (c << 5) + kbB;
      #pragma unroll
      for (int j = 0; j < NBLK; ++j) {
        const int posIdx = stride * (16 * j + nB) + t;
        const __bf16* brow = lds + posIdx * Cin;
        AFrag bf;
        bf.q[0] = *(const uint4*)(brow + kB);     // ds_load_b128
        bf.q[1] = *(const uint4*)(brow + kB + 16);
        acc[j] = __builtin_amdgcn_wmma_f32_16x16x32_bf16(
            false, a.v, false, bf.v, (short)0, acc[j], false, false);
      }
    }
  }

  // D layout: lane col n=lane&15; VGPR r -> row m = r + 8*(lane>>4)
  const int mBase = (lane >> 4) * 8;
  #pragma unroll
  for (int j = 0; j < NBLK; ++j) {
    const int l = l0 + 16 * j + nB;
    #pragma unroll
    for (int r = 0; r < 8; ++r) {
      const int co = co0 + mBase + r;
      if (co < Cout) {
        float val = acc[j][r] + bias[co];
        if (relu && val < 0.f) val = 0.f;
        const size_t off = ((size_t)b * Cout + co) * Lout + l;
        if (Yb) Yb[off] = (__bf16)val;
        if (Yf) Yf[off] = val;
      }
    }
  }
}

// Prepack fp32 W[Cout][Cin][K] -> bf16 Wp[K][CoutPad][Cin], zero pad rows.
__global__ void pack_w_kernel(const float* __restrict__ W, __bf16* __restrict__ Wp,
                              int Cout, int Cin, int K, int CoutPad)
{
  const int idx = blockIdx.x * 256 + threadIdx.x;
  const int total = K * CoutPad * Cin;
  if (idx >= total) return;
  const int ci = idx % Cin;
  const int r  = idx / Cin;
  const int co = r % CoutPad;
  const int t  = r / CoutPad;
  const float v = (co < Cout) ? W[((size_t)co * Cin + ci) * K + t] : 0.f;
  Wp[idx] = (__bf16)v;
}

// First backbone layer: Cin=1, Cout=32, k=3, stride=2, ReLU.  VALU kernel.
__global__ void bb0_kernel(const float* __restrict__ x, const float* __restrict__ w,
                           const float* __restrict__ bias, __bf16* __restrict__ y)
{
  const int idx = blockIdx.x * 256 + threadIdx.x;   // total = 16*32*8192
  const int l  = idx & 8191;
  const int r  = idx >> 13;
  const int co = r & 31;
  const int b  = r >> 5;
  if (b >= BATCH) return;
  const float* xb = x + (size_t)b * 16384;
  float acc = bias[co];
  const int p = 2 * l - 1;
  #pragma unroll
  for (int t = 0; t < 3; ++t) {
    const int gp = p + t;
    if (gp >= 0 && gp < 16384) acc += w[co * 3 + t] * xb[gp];
  }
  if (acc < 0.f) acc = 0.f;
  y[idx] = (__bf16)acc;
}

// dst[b][c][l] += src[b][c][l/2]   (nearest x2 upsample-add, bf16)
__global__ void addup_kernel(__bf16* __restrict__ dst, const __bf16* __restrict__ src,
                             int total, int Ld)
{
  const int idx = blockIdx.x * 256 + threadIdx.x;
  if (idx >= total) return;
  const int l = idx % Ld;
  const int bc = idx / Ld;
  const float v = (float)dst[idx] + (float)src[(size_t)bc * (Ld >> 1) + (l >> 1)];
  dst[idx] = (__bf16)v;
}

// Anchor generation for one FPN level.
__global__ void anchors_kernel(float* __restrict__ out, int L, float stride,
                               float a0, float a1, float a2)
{
  const int idx = blockIdx.x * 256 + threadIdx.x;
  if (idx >= 3 * L) return;
  const int a = idx / L;
  const int i = idx % L;
  const float len = (a == 0) ? a0 : ((a == 1) ? a1 : a2);
  const float c = (i + 0.5f) * stride;
  out[2 * idx + 0] = rintf(c - 0.5f * len);
  out[2 * idx + 1] = rintf(c + 0.5f * len);
}

extern "C" void kernel_launch(void* const* d_in, const int* in_sizes, int n_in,
                              void* d_out, int out_size, void* d_ws, size_t ws_size,
                              hipStream_t stream)
{
  (void)out_size; (void)ws_size;
  // Input order: {'sequences', 'params'(insertion order)}; fall back if reversed.
  int base = 1, seqIdx = 0;
  if (in_sizes && in_sizes[0] != BATCH * 16384) { base = 0; seqIdx = n_in - 1; }
  const float* seq = (const float*)d_in[seqIdx];
  auto P = [&](int k) -> const float* { return (const float*)d_in[base + k]; };
  // k: bb_w{i}=2i bb_b{i}=2i+1 (i<5); lat_w{i}=10+4i lat_b=11+4i fout_w=12+4i
  // fout_b=13+4i; cls_w{i}=22+4i cls_b=23+4i reg_w=24+4i reg_b=25+4i;
  // clsp_w=38 clsp_b=39 regp_w=40 regp_b=41

  // --- workspace carve-out (256B aligned) ---
  size_t off = 0;
  auto abf = [&](size_t n) -> __bf16* {
    __bf16* p = (__bf16*)((char*)d_ws + off);
    off += ((n * sizeof(__bf16) + 255) & ~(size_t)255);
    return p;
  };
  __bf16* wp_bb1  = abf(3 * 64 * 32);
  __bf16* wp_bb2  = abf(3 * 128 * 64);
  __bf16* wp_bb3  = abf(3 * 256 * 128);
  __bf16* wp_bb4  = abf(3 * 512 * 256);
  __bf16* wp_lat[3]  = { abf(128 * 128), abf(128 * 256), abf(128 * 512) };
  __bf16* wp_fout[3] = { abf(3 * 128 * 128), abf(3 * 128 * 128), abf(3 * 128 * 128) };
  __bf16* wp_cls[4], *wp_reg[4];
  for (int i = 0; i < 4; ++i) { wp_cls[i] = abf(3 * 128 * 128); wp_reg[i] = abf(3 * 128 * 128); }
  __bf16* wp_clsp = abf(3 * 64 * 128);
  __bf16* wp_regp = abf(3 * 16 * 128);
  __bf16* a0  = abf((size_t)BATCH * 32 * 8192);
  __bf16* a1  = abf((size_t)BATCH * 64 * 4096);
  __bf16* c3  = abf((size_t)BATCH * 128 * 2048);
  __bf16* c4  = abf((size_t)BATCH * 256 * 1024);
  __bf16* c5  = abf((size_t)BATCH * 512 * 512);
  __bf16* p5b = abf((size_t)BATCH * 128 * 512);
  __bf16* p4b = abf((size_t)BATCH * 128 * 1024);
  __bf16* p3b = abf((size_t)BATCH * 128 * 2048);
  __bf16* fpnb[3] = { abf((size_t)BATCH * 128 * 2048),
                      abf((size_t)BATCH * 128 * 1024),
                      abf((size_t)BATCH * 128 * 512) };
  __bf16* hA = abf((size_t)BATCH * 128 * 2048);
  __bf16* hB = abf((size_t)BATCH * 128 * 2048);

  // --- output layout (flattened tuple) ---
  const int Llvl[3] = {2048, 1024, 512};
  float* out = (float*)d_out;
  size_t o = 0, o_cls[3], o_reg[3], o_anc[3], o_fpn[3];
  for (int i = 0; i < 3; ++i) { o_cls[i] = o; o += (size_t)BATCH * 60 * Llvl[i]; }
  for (int i = 0; i < 3; ++i) { o_reg[i] = o; o += (size_t)BATCH * 6  * Llvl[i]; }
  for (int i = 0; i < 3; ++i) { o_anc[i] = o; o += (size_t)3 * Llvl[i] * 2; }
  for (int i = 0; i < 3; ++i) { o_fpn[i] = o; o += (size_t)BATCH * 128 * Llvl[i]; }

  auto pack = [&](const float* W, __bf16* Wp, int Cout, int Cin, int K, int CoutPad) {
    const int total = K * CoutPad * Cin;
    pack_w_kernel<<<(total + 255) / 256, 256, 0, stream>>>(W, Wp, Cout, Cin, K, CoutPad);
  };
  auto conv = [&](const __bf16* X, const __bf16* Wp, const float* bias,
                  __bf16* Yb, float* Yf, int Cin, int Cout, int CoutPad,
                  int Lin, int Lout, int strd, int ntaps, int pad, int relu,
                  int nblk) {
    const int nCo = CoutPad / 16;
    const int nLt = Lout / (16 * nblk);
    const int nCoBlk = (nCo + WAVES_PER_BLK - 1) / WAVES_PER_BLK;
    const dim3 grid((unsigned)(BATCH * nCoBlk * nLt));
    if (nblk == 4) {
      conv1d_wmma_bf16<4><<<grid, dim3(128), 0, stream>>>(
          X, Wp, bias, Yb, Yf, Cin, Cout, CoutPad, Lin, Lout, strd, ntaps, pad, relu, nCo, nLt);
    } else if (nblk == 2) {
      conv1d_wmma_bf16<2><<<grid, dim3(128), 0, stream>>>(
          X, Wp, bias, Yb, Yf, Cin, Cout, CoutPad, Lin, Lout, strd, ntaps, pad, relu, nCo, nLt);
    } else {
      conv1d_wmma_bf16<1><<<grid, dim3(128), 0, stream>>>(
          X, Wp, bias, Yb, Yf, Cin, Cout, CoutPad, Lin, Lout, strd, ntaps, pad, relu, nCo, nLt);
    }
  };

  // --- weight prepack ---
  pack(P(2), wp_bb1, 64, 32, 3, 64);
  pack(P(4), wp_bb2, 128, 64, 3, 128);
  pack(P(6), wp_bb3, 256, 128, 3, 256);
  pack(P(8), wp_bb4, 512, 256, 3, 512);
  pack(P(10), wp_lat[0], 128, 128, 1, 128);
  pack(P(14), wp_lat[1], 128, 256, 1, 128);
  pack(P(18), wp_lat[2], 128, 512, 1, 128);
  pack(P(12), wp_fout[0], 128, 128, 3, 128);
  pack(P(16), wp_fout[1], 128, 128, 3, 128);
  pack(P(20), wp_fout[2], 128, 128, 3, 128);
  for (int i = 0; i < 4; ++i) {
    pack(P(22 + 4 * i), wp_cls[i], 128, 128, 3, 128);
    pack(P(24 + 4 * i), wp_reg[i], 128, 128, 3, 128);
  }
  pack(P(38), wp_clsp, 60, 128, 3, 64);
  pack(P(40), wp_regp, 6, 128, 3, 16);

  // --- backbone ---  (stride-2 convs: NBLK=2 -> LDS window 65*Cin <= 16640)
  bb0_kernel<<<(BATCH * 32 * 8192) / 256, 256, 0, stream>>>(seq, P(0), P(1), a0);
  conv(a0, wp_bb1, P(3), a1, nullptr, 32, 64, 64, 8192, 4096, 2, 3, 1, 1, 2);
  conv(a1, wp_bb2, P(5), c3, nullptr, 64, 128, 128, 4096, 2048, 2, 3, 1, 1, 2);
  conv(c3, wp_bb3, P(7), c4, nullptr, 128, 256, 256, 2048, 1024, 2, 3, 1, 1, 2);
  conv(c4, wp_bb4, P(9), c5, nullptr, 256, 512, 512, 1024, 512, 2, 3, 1, 1, 2);

  // --- FPN top-down ---  (laterals k=1: NBLK=2 -> 32*Cin <= 16384)
  conv(c5, wp_lat[2], P(19), p5b, nullptr, 512, 128, 128, 512, 512, 1, 1, 0, 0, 2);
  conv(c4, wp_lat[1], P(15), p4b, nullptr, 256, 128, 128, 1024, 1024, 1, 1, 0, 0, 2);
  addup_kernel<<<(BATCH * 128 * 1024 + 255) / 256, 256, 0, stream>>>(
      p4b, p5b, BATCH * 128 * 1024, 1024);
  conv(c3, wp_lat[0], P(11), p3b, nullptr, 128, 128, 128, 2048, 2048, 1, 1, 0, 0, 2);
  addup_kernel<<<(BATCH * 128 * 2048 + 255) / 256, 256, 0, stream>>>(
      p3b, p4b, BATCH * 128 * 2048, 2048);
  // fout convs (stride 1, Cin=128): NBLK=4 -> 66*128 = 8448
  conv(p3b, wp_fout[0], P(13), fpnb[0], out + o_fpn[0], 128, 128, 128, 2048, 2048, 1, 3, 1, 0, 4);
  conv(p4b, wp_fout[1], P(17), fpnb[1], out + o_fpn[1], 128, 128, 128, 1024, 1024, 1, 3, 1, 0, 4);
  conv(p5b, wp_fout[2], P(21), fpnb[2], out + o_fpn[2], 128, 128, 128,  512,  512, 1, 3, 1, 0, 4);

  // --- shared heads per level ---  (all stride 1, Cin=128: NBLK=4)
  __bf16* pp[4] = { hA, hB, hA, hB };
  for (int lvl = 0; lvl < 3; ++lvl) {
    const int Lc = Llvl[lvl];
    const __bf16* cur = fpnb[lvl];
    for (int i = 0; i < 4; ++i) {
      conv(cur, wp_cls[i], P(23 + 4 * i), pp[i], nullptr, 128, 128, 128, Lc, Lc, 1, 3, 1, 1, 4);
      cur = pp[i];
    }
    conv(cur, wp_clsp, P(39), nullptr, out + o_cls[lvl], 128, 60, 64, Lc, Lc, 1, 3, 1, 0, 4);
    cur = fpnb[lvl];
    for (int i = 0; i < 4; ++i) {
      conv(cur, wp_reg[i], P(25 + 4 * i), pp[i], nullptr, 128, 128, 128, Lc, Lc, 1, 3, 1, 1, 4);
      cur = pp[i];
    }
    conv(cur, wp_regp, P(41), nullptr, out + o_reg[lvl], 128, 6, 16, Lc, Lc, 1, 3, 1, 0, 4);
  }

  // --- anchors ---
  const float astr[3] = {8.f, 16.f, 32.f};
  const float alen[3][3] = {{16.f, 32.f, 64.f}, {64.f, 128.f, 256.f}, {256.f, 512.f, 1024.f}};
  for (int lvl = 0; lvl < 3; ++lvl) {
    const int n = 3 * Llvl[lvl];
    anchors_kernel<<<(n + 255) / 256, 256, 0, stream>>>(
        out + o_anc[lvl], Llvl[lvl], astr[lvl],
        alen[lvl][0], alen[lvl][1], alen[lvl][2]);
  }
}
